// STGNNOneStep_29978871726307
// MI455X (gfx1250) — matrix operand order repaired and checked by reading, not compiled, not run on hardware
//
#include <hip/hip_runtime.h>
#include <math.h>

// ---------------------------------------------------------------------------
// STGNN one-step forward for MI455X (gfx1250), wave32, WMMA f32_16x16x32_f16.
//   B=8, A=50, T=50, HID=64, HEADS=4, HC=16
// 3 launches:
//   gat_layer_kernel (x -> H)   grid = B*T graphs
//   gat_layer_kernel (H -> H)   in-place, all reads staged to LDS first
//   tcn_head_kernel  (H -> out) grid = B*A samples, full TCN+head fused
// All B-side WMMA operands stored transposed in LDS ([N][K]) so fragment
// loads are 2x ds_load_b128; D stores transposed so they are ds_store_b128.
// Dual-tile accumulators per wave hide WMMA->WMMA RAW hazards.
// Workspace: H = [B][T][A][64] f32 = 5.12 MB in d_ws.
// ---------------------------------------------------------------------------

typedef __attribute__((ext_vector_type(16))) _Float16 v16h;
typedef __attribute__((ext_vector_type(8)))  _Float16 v8h;
typedef __attribute__((ext_vector_type(4)))  _Float16 v4h;
typedef __attribute__((ext_vector_type(8)))  float    v8f;
typedef __attribute__((ext_vector_type(4)))  float    v4f;

#define NA     50
#define NT     50
#define NB     8
#define HIDN   64
#define NHEADS 4
#define HC     16
#define THREADS 256
#define NWAVES  8

#define WMMA_F16(a, b, c) \
  __builtin_amdgcn_wmma_f32_16x16x32_f16(false, (a), false, (b), (short)0, (c), false, false)

__device__ __forceinline__ v16h cat8(v8h lo, v8h hi) {
  return __builtin_shufflevector(lo, hi, 0, 1, 2, 3, 4, 5, 6, 7,
                                 8, 9, 10, 11, 12, 13, 14, 15);
}

// A fragment (16x32 f16) from row-major LDS [row][k], ld in halves.
// lanes 0-15: row=lane, elems 0..15 -> K 0..15
// lanes 16-31: row=lane-16, elems -> K 16..31
// -> one 32B run per lane.
__device__ __forceinline__ v16h frag_a_row(const _Float16* p, int ld) {
  const int lane = threadIdx.x & 31;
  const _Float16* r = p + (lane & 15) * ld + ((lane < 16) ? 0 : 16);
  return cat8(*(const v8h*)r, *(const v8h*)(r + 8));
}

// B fragment (32x16 f16) from TRANSPOSED LDS [col][k], ld in halves.
// lane col = lane&15; elems e -> K = (lane<16?0:16)+e  -> one 32B run.
__device__ __forceinline__ v16h frag_b_colT(const _Float16* p, int ld) {
  const int lane = threadIdx.x & 31;
  const _Float16* r = p + (lane & 15) * ld + ((lane < 16) ? 0 : 16);
  return cat8(*(const v8h*)r, *(const v8h*)(r + 8));
}

// TCN conv B fragment: input stored transposed f16 [t][ci] (ld=64);
// per-lane column = causally shifted time tt (zero fragment if tt<0).
__device__ __forceinline__ v16h frag_bT_shift(const _Float16* sT, int cibase, int tt) {
  const int lane = threadIdx.x & 31;
  if (tt >= 0) {
    const _Float16* r = sT + tt * 64 + cibase + ((lane < 16) ? 0 : 16);
    return cat8(*(const v8h*)r, *(const v8h*)(r + 8));
  }
  v16h z = {};
  return z;
}

// ---------------------------------------------------------------------------
// One dense GATv2 layer over a complete graph of A=50 nodes.
// mode 0: src = x [B*A, T, 6]   (Fin=6, zero-padded to K=64)
// mode 1: src = H [B,T,A,64]    (in-place safe: staged to LDS first)
// Writes relu(out + bias_out) to Hout[b][t][a][c].
// ---------------------------------------------------------------------------
__global__ void __launch_bounds__(THREADS)
gat_layer_kernel(const float* src, int mode,
                 const float* Wl, const float* bl,
                 const float* Wr, const float* br,
                 const float* att, const float* bias_out,
                 float* Hout)
{
  __shared__ alignas(16) union {
    struct { _Float16 xin[64 * 64]; _Float16 wT[64 * 64]; } p1;  // 16 KB
    float out32[64 * 64];                                        // 16 KB
  } u;
  __shared__ alignas(16) _Float16 glT[64 * 64];    // 8 KB: (Wl x + bl)^T [c][a]
  __shared__ alignas(16) _Float16 grT[64 * 64];    // 8 KB: (Wr x + br)^T [c][a]
  __shared__ alignas(16) float    eh[NA * 52];     // 10.4 KB scores
  __shared__ alignas(16) _Float16 alphaA[64 * 64]; // 8 KB softmax head 2p
  __shared__ alignas(16) _Float16 alphaB[64 * 64]; // 8 KB softmax head 2p+1
  __shared__ float att_s[64];

  const int tid  = threadIdx.x;
  const int wid  = tid >> 5;
  const int lane = tid & 31;
  const int bt   = blockIdx.x;
  const int b    = bt / NT;
  const int t    = bt - b * NT;

  if (tid < 64) att_s[tid] = att[tid];

  // ---- phase 0: stage node features, rows a<50 / cols f, zero padded
  if (mode == 0) {
    for (int idx = tid; idx < 64 * 64; idx += THREADS) {
      int a = idx >> 6, f = idx & 63;
      float v = (a < NA && f < 6) ? src[((b * NA + a) * NT + t) * 6 + f] : 0.f;
      u.p1.xin[idx] = (_Float16)v;
    }
  } else {
    for (int idx = tid * 4; idx < 64 * 64; idx += THREADS * 4) {
      int a = idx >> 6, c = idx & 63;
      v4f v = {0.f, 0.f, 0.f, 0.f};
      if (a < NA) v = *(const v4f*)&src[((b * NT + t) * NA + a) * HIDN + c];
      v4h hv;
#pragma unroll
      for (int q = 0; q < 4; ++q) hv[q] = (_Float16)v[q];
      *(v4h*)&u.p1.xin[idx] = hv;
    }
  }
  const int Fin = (mode == 0) ? 6 : HIDN;

  // ---- phase 1: glT = (x@Wl + bl)^T, grT = (x@Wr + br)^T (dual-tile WMMA)
  for (int pass = 0; pass < 2; ++pass) {
    const float* Wp = pass ? Wr : Wl;
    const float* bp = pass ? br : bl;
    _Float16*  dstT = pass ? grT : glT;
    __syncthreads();
    for (int idx = tid; idx < 64 * 64; idx += THREADS) {
      int kf = idx >> 6, c = idx & 63;
      u.p1.wT[c * 64 + kf] = (_Float16)((kf < Fin) ? Wp[idx] : 0.f);
    }
    __syncthreads();
    {
      const int t1  = wid + NWAVES;
      const int ma0 = (wid >> 2) * 16, na0 = (wid & 3) * 16;
      const int ma1 = (t1 >> 2) * 16,  na1 = (t1 & 3) * 16;
      v8f acc0 = {}, acc1 = {};
#pragma unroll
      for (int kk = 0; kk < 64; kk += 32) {
        v16h a0 = frag_a_row(&u.p1.xin[ma0 * 64 + kk], 64);
        v16h b0 = frag_b_colT(&u.p1.wT[na0 * 64 + kk], 64);
        v16h a1 = frag_a_row(&u.p1.xin[ma1 * 64 + kk], 64);
        v16h b1 = frag_b_colT(&u.p1.wT[na1 * 64 + kk], 64);
        acc0 = WMMA_F16(a0, b0, acc0);   // independent pair: no RAW stall
        acc1 = WMMA_F16(a1, b1, acc1);
      }
      const int ro = (lane < 16) ? 0 : 8;
      {
        int col = na0 + (lane & 15);
        float bv = bp[col];
        v8h d;
#pragma unroll
        for (int r = 0; r < 8; ++r) d[r] = (_Float16)(acc0[r] + bv);
        *(v8h*)&dstT[col * 64 + ma0 + ro] = d;   // ds_store_b128
      }
      {
        int col = na1 + (lane & 15);
        float bv = bp[col];
        v8h d;
#pragma unroll
        for (int r = 0; r < 8; ++r) d[r] = (_Float16)(acc1[r] + bv);
        *(v8h*)&dstT[col * 64 + ma1 + ro] = d;
      }
    }
  }
  __syncthreads();

  // ---- phase 2: heads processed in pairs so aggregation uses all 8 waves
  for (int p2 = 0; p2 < 2; ++p2) {
    for (int hh = 0; hh < 2; ++hh) {
      const int h = p2 * 2 + hh;
      _Float16* alph = hh ? alphaB : alphaA;
      // scores e[i][j] = att_h . leaky_relu(gl_j + gr_i), diag masked
      for (int idx = tid; idx < NA * NA; idx += THREADS) {
        int i = idx / NA, j = idx - i * NA;
        float acc;
        if (i != j) {
          acc = 0.f;
#pragma unroll
          for (int c = 0; c < HC; ++c) {
            float s = (float)glT[(h * HC + c) * 64 + j] + (float)grT[(h * HC + c) * 64 + i];
            s = (s > 0.f) ? s : 0.2f * s;
            acc += att_s[h * HC + c] * s;
          }
        } else {
          acc = -3.0e38f;  // finfo.min mask -> exp()==0
        }
        eh[i * 52 + j] = acc;
      }
      __syncthreads();
      // row softmax; padded j-columns MUST be zero (padded rows of gl = bias)
      for (int i = tid; i < NA; i += THREADS) {
        float mx = -3.4e38f;
        for (int j = 0; j < NA; ++j) mx = fmaxf(mx, eh[i * 52 + j]);
        float ssum = 0.f;
        for (int j = 0; j < NA; ++j) {
          float ev = expf(eh[i * 52 + j] - mx);
          eh[i * 52 + j] = ev;
          ssum += ev;
        }
        float inv = 1.f / ssum;
        for (int j = 0; j < NA; ++j) alph[i * 64 + j] = (_Float16)(eh[i * 52 + j] * inv);
        for (int j = NA; j < 64; ++j) alph[i * 64 + j] = (_Float16)0.f;
      }
      __syncthreads();
    }
    // aggregation: waves 0-3 -> head 2p, waves 4-7 -> head 2p+1
    {
      const int h2 = p2 * 2 + (wid >> 2);
      const _Float16* alph = (wid < 4) ? alphaA : alphaB;
      const int m0 = (wid & 3) * 16;
      v8f acc = {};
#pragma unroll
      for (int kk = 0; kk < 64; kk += 32) {
        v16h af = frag_a_row(&alph[m0 * 64 + kk], 64);
        v16h bf = frag_b_colT(&glT[(h2 * HC) * 64 + kk], 64);
        acc = WMMA_F16(af, bf, acc);
      }
      const int colc = h2 * HC + (lane & 15);
      const int mbr  = m0 + ((lane < 16) ? 0 : 8);
#pragma unroll
      for (int r = 0; r < 8; ++r)
        u.out32[(mbr + r) * 64 + colc] = acc[r];
    }
    __syncthreads();
  }

  // ---- phase 3: relu(out + bias_out) -> H  (b128 both sides)
  for (int idx = tid * 4; idx < NA * HIDN; idx += THREADS * 4) {
    int a = idx >> 6, c = idx & 63;
    v4f v  = *(const v4f*)&u.out32[idx];
    v4f bb = *(const v4f*)&bias_out[c];
    v4f r2;
#pragma unroll
    for (int q = 0; q < 4; ++q) {
      float z = v[q] + bb[q];
      r2[q] = (z > 0.f) ? z : 0.f;
    }
    *(v4f*)&Hout[((b * NT + t) * NA + a) * HIDN + c] = r2;
  }
}

// ---------------------------------------------------------------------------
// Fused TCN (3 residual blocks, dilations 1/2/4, 2 causal convs each) + head.
// One block per sample n = b*A + a; whole chain sample-local in LDS.
// f32 [ci][t] residual buffer + f16 [t][ci] shadow for WMMA B fragments.
// ---------------------------------------------------------------------------
__global__ void __launch_bounds__(THREADS)
tcn_head_kernel(const float* H,
                const float* w1, const float* b1,
                const float* w2, const float* b2,
                const float* hw1, const float* hb1,
                const float* hw2, const float* hb2,
                float* out)
{
  __shared__ alignas(16) float    hbuf[64 * 64];   // 16 KB f32 [ci][t]
  __shared__ alignas(16) _Float16 hT[64 * 64];     //  8 KB f16 [t][ci]
  __shared__ alignas(16) _Float16 oT[64 * 64];     //  8 KB f16 [t][ci] conv1 out
  __shared__ alignas(16) _Float16 wbuf[64 * 192];  // 24 KB W f16 [co][k*64+ci]
  __shared__ float tmp32[32];

  const int tid  = threadIdx.x;
  const int wid  = tid >> 5;
  const int lane = tid & 31;
  const int n    = blockIdx.x;
  const int b    = n / NA;
  const int a    = n - b * NA;

  // stage h[n]; pad t>=50 with zeros (keeps causal pad clean through chain)
  for (int idx = tid * 4; idx < 64 * 64; idx += THREADS * 4) {
    int t = idx >> 6, c = idx & 63;
    v4f v = {0.f, 0.f, 0.f, 0.f};
    if (t < NT) v = *(const v4f*)&H[((b * NT + t) * NA + a) * HIDN + c];
    hbuf[(c + 0) * 64 + t] = v[0];
    hbuf[(c + 1) * 64 + t] = v[1];
    hbuf[(c + 2) * 64 + t] = v[2];
    hbuf[(c + 3) * 64 + t] = v[3];
    v4h hv;
#pragma unroll
    for (int q = 0; q < 4; ++q) hv[q] = (_Float16)v[q];
    *(v4h*)&hT[idx] = hv;
  }

  for (int blk = 0; blk < 3; ++blk) {
    const int d = 1 << blk;
    for (int conv = 0; conv < 2; ++conv) {
      const float* wsrc = (conv ? w2 : w1) + blk * (64 * 64 * 3);
      const float* bsrc = (conv ? b2 : b1) + blk * 64;
      const _Float16* sT = conv ? oT : hT;
      if (tid == 0 && !(blk == 2 && conv == 1))
        __builtin_prefetch(wsrc + 64 * 64 * 3, 0, 1);  // this conv's weights for the next block
      __syncthreads();
      // W [co][ci][k] f32 -> f16 LDS [co][k*64+ci] (no integer divisions)
      for (int s = tid; s < 64 * 64; s += THREADS) {
        int co = s >> 6, ci = s & 63;
#pragma unroll
        for (int k = 0; k < 3; ++k)
          wbuf[co * 192 + k * 64 + ci] = (_Float16)wsrc[co * 192 + ci * 3 + k];
      }
      __syncthreads();
      // out[co][t] = sum_k sum_ci W_k[co][ci] * src[ci][t-(2-k)d], dual tiles
      {
        const int t1  = wid + NWAVES;
        const int m0a = (wid >> 2) * 16, n0a = (wid & 3) * 16;
        const int m0b = (t1 >> 2) * 16,  n0b = (t1 & 3) * 16;
        const int colA = n0a + (lane & 15);
        const int colB = n0b + (lane & 15);
        v8f acc0 = {}, acc1 = {};
#pragma unroll
        for (int k = 0; k < 3; ++k) {
          const int ttA = colA - (2 - k) * d;
          const int ttB = colB - (2 - k) * d;
#pragma unroll
          for (int kk = 0; kk < 64; kk += 32) {
            v16h a0 = frag_a_row(&wbuf[m0a * 192 + k * 64 + kk], 192);
            v16h b0 = frag_bT_shift(sT, kk, ttA);
            acc0 = WMMA_F16(a0, b0, acc0);
            v16h a1 = frag_a_row(&wbuf[m0b * 192 + k * 64 + kk], 192);
            v16h b1 = frag_bT_shift(sT, kk, ttB);
            acc1 = WMMA_F16(a1, b1, acc1);
          }
        }
        const int ro = (lane < 16) ? 0 : 8;
        if (conv == 0) {
          // o = relu(conv1 + b): contiguous f16 store into oT
          {
            const int mb = m0a + ro;
            v4f bv0 = *(const v4f*)&bsrc[mb], bv1 = *(const v4f*)&bsrc[mb + 4];
            v8h dd;
#pragma unroll
            for (int r = 0; r < 8; ++r) {
              float z = acc0[r] + ((r < 4) ? bv0[r & 3] : bv1[r & 3]);
              dd[r] = (_Float16)((z > 0.f) ? z : 0.f);
            }
            *(v8h*)&oT[colA * 64 + mb] = dd;
          }
          {
            const int mb = m0b + ro;
            v4f bv0 = *(const v4f*)&bsrc[mb], bv1 = *(const v4f*)&bsrc[mb + 4];
            v8h dd;
#pragma unroll
            for (int r = 0; r < 8; ++r) {
              float z = acc1[r] + ((r < 4) ? bv0[r & 3] : bv1[r & 3]);
              dd[r] = (_Float16)((z > 0.f) ? z : 0.f);
            }
            *(v8h*)&oT[colB * 64 + mb] = dd;
          }
        } else {
          // h = relu(conv2 + b) + h : update f32 residual + f16 shadow
          if (colA < NT) {
            const int mb = m0a + ro;
            v4f bv0 = *(const v4f*)&bsrc[mb], bv1 = *(const v4f*)&bsrc[mb + 4];
            v8h dd;
#pragma unroll
            for (int r = 0; r < 8; ++r) {
              float z = acc0[r] + ((r < 4) ? bv0[r & 3] : bv1[r & 3]);
              z = (z > 0.f) ? z : 0.f;
              float nh = hbuf[(mb + r) * 64 + colA] + z;
              hbuf[(mb + r) * 64 + colA] = nh;
              dd[r] = (_Float16)nh;
            }
            *(v8h*)&hT[colA * 64 + mb] = dd;
          }
          if (colB < NT) {
            const int mb = m0b + ro;
            v4f bv0 = *(const v4f*)&bsrc[mb], bv1 = *(const v4f*)&bsrc[mb + 4];
            v8h dd;
#pragma unroll
            for (int r = 0; r < 8; ++r) {
              float z = acc1[r] + ((r < 4) ? bv0[r & 3] : bv1[r & 3]);
              z = (z > 0.f) ? z : 0.f;
              float nh = hbuf[(mb + r) * 64 + colB] + z;
              hbuf[(mb + r) * 64 + colB] = nh;
              dd[r] = (_Float16)nh;
            }
            *(v8h*)&hT[colB * 64 + mb] = dd;
          }
        }
      }
    }
  }
  __syncthreads();

  // head on t = T-1: silu(h @ W1^T + b1) @ W2^T + b2
  if (tid < 32) {
    float s = hb1[tid];
    for (int k = 0; k < 64; ++k) s += hw1[tid * 64 + k] * hbuf[k * 64 + (NT - 1)];
    tmp32[tid] = s / (1.f + expf(-s));
  }
  __syncthreads();
  if (tid < 6) {
    float o = hb2[tid];
    for (int k = 0; k < 32; ++k) o += hw2[tid * 32 + k] * tmp32[k];
    out[n * 6 + tid] = o;
  }
}

// ---------------------------------------------------------------------------
extern "C" void kernel_launch(void* const* d_in, const int* in_sizes, int n_in,
                              void* d_out, int out_size, void* d_ws, size_t ws_size,
                              hipStream_t stream) {
  (void)in_sizes; (void)n_in; (void)out_size; (void)ws_size;
  const float* x     = (const float*)d_in[0];
  const float* g1Wl  = (const float*)d_in[1];
  const float* g1bl  = (const float*)d_in[2];
  const float* g1Wr  = (const float*)d_in[3];
  const float* g1br  = (const float*)d_in[4];
  const float* g1att = (const float*)d_in[5];
  const float* g1bs  = (const float*)d_in[6];
  const float* g2Wl  = (const float*)d_in[7];
  const float* g2bl  = (const float*)d_in[8];
  const float* g2Wr  = (const float*)d_in[9];
  const float* g2br  = (const float*)d_in[10];
  const float* g2att = (const float*)d_in[11];
  const float* g2bs  = (const float*)d_in[12];
  const float* tw1   = (const float*)d_in[13];
  const float* tb1   = (const float*)d_in[14];
  const float* tw2   = (const float*)d_in[15];
  const float* tb2   = (const float*)d_in[16];
  const float* hw1   = (const float*)d_in[17];
  const float* hb1   = (const float*)d_in[18];
  const float* hw2   = (const float*)d_in[19];
  const float* hb2   = (const float*)d_in[20];

  float* H = (float*)d_ws;  // [B][T][A][64] f32 = 5.12 MB

  gat_layer_kernel<<<dim3(NB * NT), dim3(THREADS), 0, stream>>>(
      x, 0, g1Wl, g1bl, g1Wr, g1br, g1att, g1bs, H);
  gat_layer_kernel<<<dim3(NB * NT), dim3(THREADS), 0, stream>>>(
      H, 1, g2Wl, g2bl, g2Wr, g2br, g2att, g2bs, H);
  tcn_head_kernel<<<dim3(NB * NA), dim3(THREADS), 0, stream>>>(
      H, tw1, tb1, tw2, tb2, hw1, hb1, hw2, hb2, (float*)d_out);
}
